// Mamba2Block_53446573031683
// MI455X (gfx1250) — matrix operand order repaired and checked by reading, not compiled, not run on hardware
//
#include <hip/hip_runtime.h>
#include <hip/hip_bf16.h>
#include <math.h>

// ---------------- model constants ----------------
#define D_MODEL   256
#define D_INNER   512
#define HEADDIM   64
#define NHEADS    8
#define D_STATE   64
#define D_CONV    4
#define CONV_DIM  640            // D_INNER + 2*D_STATE
#define D_IN_PROJ 1552           // z | xBC | ... | dt (last NHEADS cols)
#define BATCH     2
#define SEQLEN    2048
#define BL        (BATCH*SEQLEN) // 4096
#define EPS       1e-5f

// ---------------- gfx1250 async-to-LDS availability ----------------
#if defined(__has_builtin)
#if __has_builtin(__builtin_amdgcn_global_load_async_to_lds_b128) && \
    __has_builtin(__builtin_amdgcn_global_load_async_to_lds_b32)  && \
    __has_builtin(__builtin_amdgcn_s_wait_asynccnt)
#define HAVE_ASYNC_LDS 1
#endif
#endif

typedef int v4i __attribute__((ext_vector_type(4)));

#ifdef HAVE_ASYNC_LDS
typedef __attribute__((address_space(1))) v4i* g_v4i_p;
typedef __attribute__((address_space(3))) v4i* l_v4i_p;
typedef __attribute__((address_space(1))) int* g_i32_p;
typedef __attribute__((address_space(3))) int* l_i32_p;
__device__ __forceinline__ void async_copy_b128(const void* gsrc, void* ldst) {
    __builtin_amdgcn_global_load_async_to_lds_b128((g_v4i_p)gsrc, (l_v4i_p)ldst, 0, 0);
}
__device__ __forceinline__ void async_copy_b32(const void* gsrc, void* ldst) {
    __builtin_amdgcn_global_load_async_to_lds_b32((g_i32_p)gsrc, (l_i32_p)ldst, 0, 0);
}
__device__ __forceinline__ void async_wait_all() {
    __builtin_amdgcn_s_wait_asynccnt(0);
}
#else
__device__ __forceinline__ void async_copy_b128(const void* gsrc, void* ldst) {
    *(float4*)ldst = *(const float4*)gsrc;
}
__device__ __forceinline__ void async_copy_b32(const void* gsrc, void* ldst) {
    *(float*)ldst = *(const float*)gsrc;
}
__device__ __forceinline__ void async_wait_all() {}
#endif

// ---------------- WMMA types ----------------
typedef __attribute__((ext_vector_type(16))) __bf16 v16bf;
typedef __attribute__((ext_vector_type(8)))  float  v8f;

union ABFrag {
    unsigned int u[8];
    v16bf        v;
};

__device__ __forceinline__ unsigned short f32_to_bf16_rne(float f) {
    unsigned int u = __float_as_uint(f);
    if ((u & 0x7F800000u) == 0x7F800000u)     // inf / nan: truncate
        return (unsigned short)(u >> 16);
    unsigned int r = u + 0x7FFFu + ((u >> 16) & 1u);
    return (unsigned short)(r >> 16);
}

__device__ __forceinline__ float silu_f(float x) {
    return x * (1.0f / (1.0f + __expf(-x)));
}

// ---------------- 1) LayerNorm + residual copy ----------------
__global__ void ln_kernel(const float* __restrict__ hidden,
                          const float* __restrict__ nw,
                          const float* __restrict__ nb,
                          unsigned short* __restrict__ xln,   // bf16 (BL x 256)
                          float* __restrict__ residual_out) {
    __shared__ float red[256];
    int row = blockIdx.x, t = threadIdx.x;
    float v = hidden[row * D_MODEL + t];
    residual_out[row * D_MODEL + t] = v;

    red[t] = v; __syncthreads();
    for (int s = 128; s > 0; s >>= 1) { if (t < s) red[t] += red[t + s]; __syncthreads(); }
    float mu = red[0] * (1.0f / D_MODEL);
    __syncthreads();
    float d = v - mu;
    red[t] = d * d; __syncthreads();
    for (int s = 128; s > 0; s >>= 1) { if (t < s) red[t] += red[t + s]; __syncthreads(); }
    float var = red[0] * (1.0f / D_MODEL);
    float xn  = d * rsqrtf(var + EPS) * nw[t] + nb[t];
    xln[row * D_MODEL + t] = f32_to_bf16_rne(xn);
}

// ---------------- 2) pack weight (K x N f32) -> K-pair-major bf16x2 ----------------
__global__ void pack_w_kernel(const float* __restrict__ W, unsigned int* __restrict__ dst,
                              int K, int N) {
    int idx = blockIdx.x * blockDim.x + threadIdx.x;
    int tot = (K >> 1) * N;
    if (idx >= tot) return;
    int pk = idx / N, n = idx - pk * N;
    unsigned int lo = f32_to_bf16_rne(W[(2 * pk)     * N + n]);
    unsigned int hi = f32_to_bf16_rne(W[(2 * pk + 1) * N + n]);
    dst[pk * N + n] = lo | (hi << 16);
}

// ---------------- 3) WMMA bf16 GEMM:  C[M,N] = A[M,K] * B[K,N] ----------------
// Each wave computes a 64x16 tile: 4 M-subtiles share one B fragment per K-step
// (amortizes the strided B loads 4x). 8 waves / 256-thread block.
// Fragment layouts per CDNA5 ISA 7.12.2 (bf16 16x16x32):
//   A: lane {M=lane&15, half=lane>>4}; VGPR v<4: K=2v+8h ; v>=4: K=16+2(v-4)+8h
//   B: lane {N=lane&15, half};         VGPR v<4: K=2v+8h ; v>=4: K=16+2(v-4)+8h
//   C: VGPR r holds M=r+8h, N=lane&15
template <int M, int K, int N>
__global__ void wmma_gemm_kernel(const unsigned int* __restrict__ A32, // M x K/2
                                 const unsigned int* __restrict__ B32, // K/2 x N
                                 float* __restrict__ C) {
    const int NT = N / 16;
    int wave = threadIdx.x >> 5;
    int lane = threadIdx.x & 31;
    int tile = blockIdx.x * 8 + wave;
    int mi = tile / NT, ni = tile - mi * NT;      // mi: 64-row block, ni: 16-col block
    int half = lane >> 4;
    int nl   = lane & 15;

    const unsigned int* aRow[4];
#pragma unroll
    for (int s = 0; s < 4; ++s)
        aRow[s] = A32 + (size_t)(mi * 64 + s * 16 + nl) * (K / 2);
    const unsigned int* bCol = B32 + (ni * 16 + nl);

    int apk[8], bpk[8];
#pragma unroll
    for (int v = 0; v < 8; ++v) {
        apk[v] = (v < 4) ? (v + 4 * half) : (8 + (v - 4) + 4 * half);
        bpk[v] = apk[v];
    }

    v8f c0 = {}, c1 = {}, c2 = {}, c3 = {};
    for (int kk = 0; kk < K; kk += 32) {
        int pkb = kk >> 1;
        if (kk + 32 < K) {
            __builtin_prefetch((const void*)(aRow[0] + pkb + 16), 0, 1);
            __builtin_prefetch((const void*)(bCol + (size_t)(pkb + 16) * N), 0, 1);
        }
        ABFrag b;
#pragma unroll
        for (int v = 0; v < 8; ++v) b.u[v] = bCol[(size_t)(pkb + bpk[v]) * N];
        ABFrag a0, a1, a2, a3;
#pragma unroll
        for (int v = 0; v < 8; ++v) {
            a0.u[v] = aRow[0][pkb + apk[v]];
            a1.u[v] = aRow[1][pkb + apk[v]];
            a2.u[v] = aRow[2][pkb + apk[v]];
            a3.u[v] = aRow[3][pkb + apk[v]];
        }
        c0 = __builtin_amdgcn_wmma_f32_16x16x32_bf16(false, a0.v, false, b.v, (short)0, c0, false, false);
        c1 = __builtin_amdgcn_wmma_f32_16x16x32_bf16(false, a1.v, false, b.v, (short)0, c1, false, false);
        c2 = __builtin_amdgcn_wmma_f32_16x16x32_bf16(false, a2.v, false, b.v, (short)0, c2, false, false);
        c3 = __builtin_amdgcn_wmma_f32_16x16x32_bf16(false, a3.v, false, b.v, (short)0, c3, false, false);
    }

    int ncol = ni * 16 + nl;
#pragma unroll
    for (int s = 0; s < 4; ++s) {
        const v8f& c = (s == 0) ? c0 : (s == 1) ? c1 : (s == 2) ? c2 : c3;
        float* crow = C + (size_t)(mi * 64 + s * 16 + 8 * half) * N + ncol;
#pragma unroll
        for (int r = 0; r < 8; ++r) crow[(size_t)r * N] = c[r];
    }
}

// ---------------- 4) causal depthwise conv (K=4) + SiLU ----------------
__global__ void conv_silu_kernel(const float* __restrict__ zx,   // BL x 1552
                                 const float* __restrict__ cw,   // 640 x 4
                                 const float* __restrict__ cb,
                                 float* __restrict__ xbc) {       // BL x 640
    int idx = blockIdx.x * blockDim.x + threadIdx.x;
    int row = idx / CONV_DIM, c = idx - row * CONV_DIM;
    int b = row >> 11, l = row & (SEQLEN - 1);
    float acc = cb[c];
#pragma unroll
    for (int k = 0; k < D_CONV; ++k) {
        int ll = l + k - (D_CONV - 1);
        if (ll >= 0)
            acc += cw[c * D_CONV + k] * zx[(size_t)(b * SEQLEN + ll) * D_IN_PROJ + D_INNER + c];
    }
    xbc[(size_t)row * CONV_DIM + c] = silu_f(acc);
}

// ---------------- 5) dt softplus + dA ----------------
__global__ void dt_kernel(const float* __restrict__ zx,
                          const float* __restrict__ dt_bias,
                          const float* __restrict__ A_log,
                          float* __restrict__ dtb, float* __restrict__ dab) {
    int idx = blockIdx.x * blockDim.x + threadIdx.x;
    int row = idx >> 3, h = idx & 7;
    float raw = zx[(size_t)row * D_IN_PROJ + (D_IN_PROJ - NHEADS) + h] + dt_bias[h];
    float dt  = (raw > 20.0f) ? raw : log1pf(__expf(raw));
    float Anh = -__expf(A_log[h]);
    dtb[idx] = dt;
    dab[idx] = __expf(dt * Anh);
}

// ---------------- 6) selective scan (latency-critical sequential part) ----------------
// 16 blocks = (B,H). State S[p=64][n=64] in registers: thread t owns p=t>>2,
// n in [16*(t&3), +16). Chunks of 8 timesteps are double-buffered in LDS and
// filled with gfx1250 GLOBAL_LOAD_ASYNC_TO_LDS (ASYNCcnt) so the global-memory
// latency of chunk i+1 hides behind the recurrence math of chunk i.
#define SCAN_CH 8
__global__ void scan_kernel(const float* __restrict__ xbc,  // BL x 640
                            const float* __restrict__ dtb,  // BL x 8
                            const float* __restrict__ dab,  // BL x 8
                            const float* __restrict__ Dp_,
                            float* __restrict__ yg) {        // BL x 512
    // lBC: cols 0..63 = B, 64..127 = C (contiguous in source: cols 512..639)
    __shared__ __align__(16) float lBC[2][SCAN_CH][128];
    __shared__ __align__(16) float lX [2][SCAN_CH][HEADDIM];
    __shared__ __align__(16) float lS[2][SCAN_CH][2];   // [0]=dA, [1]=dt

    int bh = blockIdx.x;
    int b = bh >> 3, h = bh & 7;
    int t = threadIdx.x;
    int p = t >> 2, q = t & 3, n0 = q * 16;
    float Dp = Dp_[h];

    // issue stage of one chunk into buffer `buf` (384 x b128 + 16 x b32)
    auto issue_chunk = [&](int ci, int buf) {
        int base_row = b * SEQLEN + ci * SCAN_CH;
        for (int i = t; i < 400; i += 256) {
            if (i < 384) {
                int st = i / 48, j = i - st * 48;
                size_t row = (size_t)(base_row + st) * CONV_DIM;
                if (j < 32)
                    async_copy_b128(xbc + row + D_INNER + j * 4, &lBC[buf][st][j * 4]);
                else
                    async_copy_b128(xbc + row + h * HEADDIM + (j - 32) * 4,
                                    &lX[buf][st][(j - 32) * 4]);
            } else {
                int k = i - 384, st = k >> 1, w = k & 1;
                const float* src = (w ? dtb : dab) + (size_t)(base_row + st) * NHEADS + h;
                async_copy_b32(src, &lS[buf][st][w]);
            }
        }
    };

    float s[16];
#pragma unroll
    for (int i = 0; i < 16; ++i) s[i] = 0.0f;

    issue_chunk(0, 0);
    for (int ci = 0; ci < SEQLEN / SCAN_CH; ++ci) {
        int buf = ci & 1;
        async_wait_all();      // my async loads done
        __syncthreads();       // everyone's done; prior buffer free for rewrite
        if (ci + 1 < SEQLEN / SCAN_CH) issue_chunk(ci + 1, buf ^ 1);

#pragma unroll
        for (int st = 0; st < SCAN_CH; ++st) {
            float da  = lS[buf][st][0];
            float dtv = lS[buf][st][1];
            float xp  = lX[buf][st][p];
            float dx  = dtv * xp;
            float yp  = 0.0f;
#pragma unroll
            for (int i = 0; i < 16; ++i) {
                s[i] = s[i] * da + dx * lBC[buf][st][n0 + i];
                yp  += s[i] * lBC[buf][st][64 + n0 + i];
            }
            yp += __shfl_xor(yp, 1, 32);
            yp += __shfl_xor(yp, 2, 32);
            if (q == 0) {
                int row = b * SEQLEN + ci * SCAN_CH + st;
                yg[(size_t)row * D_INNER + h * HEADDIM + p] = yp + Dp * xp;
            }
        }
    }
}

// ---------------- 7) gate (silu(z)) + RMSNorm -> bf16 ----------------
__global__ void gate_rms_kernel(const float* __restrict__ zx,   // BL x 1552 (z=[:512])
                                const float* __restrict__ yg,   // BL x 512
                                const float* __restrict__ rw,
                                unsigned short* __restrict__ gb) {
    __shared__ float red[256];
    int row = blockIdx.x, t = threadIdx.x;
    size_t rz = (size_t)row * D_IN_PROJ, ry = (size_t)row * D_INNER;
    float g0 = yg[ry + t]       * silu_f(zx[rz + t]);
    float g1 = yg[ry + t + 256] * silu_f(zx[rz + t + 256]);
    red[t] = g0 * g0 + g1 * g1; __syncthreads();
    for (int s = 128; s > 0; s >>= 1) { if (t < s) red[t] += red[t + s]; __syncthreads(); }
    float r = rsqrtf(red[0] * (1.0f / D_INNER) + EPS);
    gb[ry + t]       = f32_to_bf16_rne(g0 * r * rw[t]);
    gb[ry + t + 256] = f32_to_bf16_rne(g1 * r * rw[t + 256]);
}

// ---------------- launch ----------------
extern "C" void kernel_launch(void* const* d_in, const int* in_sizes, int n_in,
                              void* d_out, int out_size, void* d_ws, size_t ws_size,
                              hipStream_t stream) {
    const float* hidden  = (const float*)d_in[0];
    const float* norm_w  = (const float*)d_in[1];
    const float* norm_b  = (const float*)d_in[2];
    const float* W_in    = (const float*)d_in[3];
    const float* conv_w  = (const float*)d_in[4];
    const float* conv_b  = (const float*)d_in[5];
    const float* dt_bias = (const float*)d_in[6];
    const float* A_log   = (const float*)d_in[7];
    const float* D_param = (const float*)d_in[8];
    const float* rms_w   = (const float*)d_in[9];
    const float* W_out   = (const float*)d_in[10];
    float* out = (float*)d_out;

    auto align256 = [](size_t x) { return (x + 255) & ~(size_t)255; };
    char* ws = (char*)d_ws;
    size_t o = 0;
    unsigned short* xln   = (unsigned short*)(ws + o); o = align256(o + (size_t)BL * D_MODEL * 2);
    unsigned int*   winP  = (unsigned int*)  (ws + o); o = align256(o + (size_t)(D_MODEL/2) * D_IN_PROJ * 4);
    unsigned int*   woutP = (unsigned int*)  (ws + o); o = align256(o + (size_t)(D_INNER/2) * D_MODEL * 4);
    float*          zx    = (float*)         (ws + o); o = align256(o + (size_t)BL * D_IN_PROJ * 4);
    float*          xbc   = (float*)         (ws + o); o = align256(o + (size_t)BL * CONV_DIM * 4);
    float*          dtb   = (float*)         (ws + o); o = align256(o + (size_t)BL * NHEADS * 4);
    float*          dab   = (float*)         (ws + o); o = align256(o + (size_t)BL * NHEADS * 4);
    float*          yg    = (float*)         (ws + o); o = align256(o + (size_t)BL * D_INNER * 4);
    unsigned short* gb    = (unsigned short*)(ws + o); o = align256(o + (size_t)BL * D_INNER * 2);
    (void)ws_size; (void)in_sizes; (void)n_in; (void)out_size;

    ln_kernel<<<BL, 256, 0, stream>>>(hidden, norm_w, norm_b, xln, out + (size_t)BL * D_MODEL);

    {
        int tot = (D_MODEL / 2) * D_IN_PROJ;
        pack_w_kernel<<<(tot + 255) / 256, 256, 0, stream>>>(W_in, winP, D_MODEL, D_IN_PROJ);
        tot = (D_INNER / 2) * D_MODEL;
        pack_w_kernel<<<(tot + 255) / 256, 256, 0, stream>>>(W_out, woutP, D_INNER, D_MODEL);
    }

    // in-projection: 64 x 97 wave-tiles of 64x16 -> 776 blocks
    wmma_gemm_kernel<BL, D_MODEL, D_IN_PROJ>
        <<<(BL / 64) * (D_IN_PROJ / 16) / 8, 256, 0, stream>>>((const unsigned int*)xln, winP, zx);

    conv_silu_kernel<<<(BL * CONV_DIM) / 256, 256, 0, stream>>>(zx, conv_w, conv_b, xbc);
    dt_kernel<<<(BL * NHEADS) / 256, 256, 0, stream>>>(zx, dt_bias, A_log, dtb, dab);

    scan_kernel<<<BATCH * NHEADS, 256, 0, stream>>>(xbc, dtb, dab, D_param, yg);

    gate_rms_kernel<<<BL, 256, 0, stream>>>(zx, yg, rms_w, gb);

    // out-projection: 64 x 16 wave-tiles of 64x16 -> 128 blocks
    wmma_gemm_kernel<BL, D_INNER, D_MODEL>
        <<<(BL / 64) * (D_MODEL / 16) / 8, 256, 0, stream>>>((const unsigned int*)gb, woutP, out);
}
